// enc_mtan_GRU_59785944760916
// MI455X (gfx1250) — matrix operand order repaired and checked by reading, not compiled, not run on hardware
//
#include <hip/hip_runtime.h>

// Problem constants
#define Bc 32
#define Sc 512
#define Rc 128
#define Dc 64
#define Ec 128
#define Hc 256
#define THc 768   // 3*H

typedef __bf16 bf16_t;
typedef __bf16 v16bf __attribute__((ext_vector_type(16)));
typedef __bf16 v8bf  __attribute__((ext_vector_type(8)));
typedef float  v8f   __attribute__((ext_vector_type(8)));

static __device__ __forceinline__ v8f wmma_bf16(v16bf a, v16bf b, v8f c) {
  return __builtin_amdgcn_wmma_f32_16x16x32_bf16(false, a, false, b, (short)0, c,
                                                 false, false);
}

// ---- Fragment loaders (contiguous k order: k = 16*(lane/16) + elem, same for A and B) ----

// A: 16x32 bf16, source row-major with leading dim lda, tile origin = A.
// lane<16: elems 0..15 = K kb..kb+15 (kb=0)
// lane>=16: kb=16
static __device__ __forceinline__ v16bf load_frag_a(const bf16_t* A, int lda) {
  int lane = threadIdx.x & 31;
  int row  = lane & 15;
  int kb   = (lane & 16) ? 16 : 0;
  const bf16_t* p = A + row * lda + kb;
  v8bf lo = *reinterpret_cast<const v8bf*>(p);
  v8bf hi = *reinterpret_cast<const v8bf*>(p + 8);
  v16bf r;
#pragma unroll
  for (int i = 0; i < 8; ++i) { r[i] = lo[i]; r[i + 8] = hi[i]; }
  return r;
}

// B: 32x16 bf16 from pre-packed fragment buffer (tile = 32 lanes x 16 elems)
static __device__ __forceinline__ v16bf load_frag_b_pack(const bf16_t* Bp) {
  int lane = threadIdx.x & 31;
  const bf16_t* p = Bp + lane * 16;
  v8bf lo = *reinterpret_cast<const v8bf*>(p);
  v8bf hi = *reinterpret_cast<const v8bf*>(p + 8);
  v16bf r;
#pragma unroll
  for (int i = 0; i < 8; ++i) { r[i] = lo[i]; r[i + 8] = hi[i]; }
  return r;
}

// B: 32x16 where source memory is [N][K] row-major (K contiguous).
// Tile origin passed as Bsrc = base + n0*ldb + k0.
static __device__ __forceinline__ v16bf load_frag_b_rowK(const bf16_t* Bsrc, int ldb) {
  int lane = threadIdx.x & 31;
  int col  = lane & 15;
  int kb   = (lane & 16) ? 16 : 0;
  const bf16_t* p = Bsrc + col * ldb + kb;
  v8bf lo = *reinterpret_cast<const v8bf*>(p);
  v8bf hi = *reinterpret_cast<const v8bf*>(p + 8);
  v16bf r;
#pragma unroll
  for (int i = 0; i < 8; ++i) { r[i] = lo[i]; r[i + 8] = hi[i]; }
  return r;
}

// ---- Kernel 0: pack W (N x K, row-major, f32) into bf16 B-fragment order ----
__global__ void pack_wT_kernel(const float* __restrict__ W, bf16_t* __restrict__ dst,
                               int N, int K) {
  int idx = blockIdx.x * blockDim.x + threadIdx.x;
  if (idx >= N * K) return;
  int n = idx / K, k = idx % K;
  int kt = k >> 5, kk = k & 31, nt = n >> 4, col = n & 15;
  int lane = col + (kk & 16);
  int e = kk & 15;
  int NT = N >> 4;
  dst[(((kt * NT + nt) << 5) + lane) * 16 + e] = (bf16_t)W[idx];
}

// ---- Kernel 1: time embeddings (bf16) ----
__global__ void embed_kernel(const float* __restrict__ ts, const float* __restrict__ qt,
                             const float* __restrict__ w_per, const float* __restrict__ b_per,
                             const float* __restrict__ w_lin, const float* __restrict__ b_lin,
                             bf16_t* __restrict__ k_emb, bf16_t* __restrict__ q_emb) {
  int idx = blockIdx.x * blockDim.x + threadIdx.x;
  const int total = (Bc * Sc + Rc) * Ec;
  if (idx >= total) return;
  int pos = idx >> 7;
  int e = idx & 127;
  float tt = (pos < Bc * Sc) ? ts[pos] : qt[pos - Bc * Sc];
  float v = (e == 0) ? (tt * w_lin[0] + b_lin[0])
                     : __sinf(tt * w_per[e - 1] + b_per[e - 1]);
  if (pos < Bc * Sc) k_emb[idx] = (bf16_t)v;
  else               q_emb[(pos - Bc * Sc) * Ec + e] = (bf16_t)v;
}

// ---- Kernel 2: pack x*mask and mask into per-batch B-fragment order ----
__global__ void pack_x_kernel(const float* __restrict__ xv, const int* __restrict__ xm,
                              bf16_t* __restrict__ xm_pack, bf16_t* __restrict__ mk_pack) {
  int idx = blockIdx.x * blockDim.x + threadIdx.x;
  if (idx >= Bc * Sc * Dc) return;
  int b = idx / (Sc * Dc);
  int rem = idx % (Sc * Dc);
  int s = rem / Dc, d = rem % Dc;
  float m = (xm[idx] != 0) ? 1.f : 0.f;
  int kt = s >> 5, kk = s & 31, nt = d >> 4, col = d & 15;
  int lane = col + (kk & 16);
  int e = kk & 15;
  int off = b * (Sc * Dc) + (((kt * (Dc / 16) + nt) << 5) + lane) * 16 + e;
  xm_pack[off] = (bf16_t)(xv[idx] * m);
  mk_pack[off] = (bf16_t)m;
}

// ---- Kernel 3: projection GEMM: C(Mx128) = A(Mx128) x Bpack + bias, bf16 out ----
__global__ void __launch_bounds__(256)
proj_kernel(const bf16_t* __restrict__ A, const bf16_t* __restrict__ Bp,
            const float* __restrict__ bias, bf16_t* __restrict__ C, int M) {
  int wave = (blockIdx.x * blockDim.x + threadIdx.x) >> 5;
  int nt = wave & 7, mt = wave >> 3;
  if (mt >= (M >> 4)) return;
  v8f acc = {};
  const bf16_t* At = A + (mt * 16) * Ec;
#pragma unroll
  for (int kt = 0; kt < 4; ++kt) {
    v16bf a = load_frag_a(At + kt * 32, Ec);
    v16bf b = load_frag_b_pack(Bp + ((kt * 8 + nt) << 9));
    acc = wmma_bf16(a, b, acc);
  }
  int lane = threadIdx.x & 31;
  int col = nt * 16 + (lane & 15);
  int rb = mt * 16 + ((lane & 16) ? 8 : 0);
  float bc = bias[col];
#pragma unroll
  for (int v = 0; v < 8; ++v) C[(rb + v) * Ec + col] = (bf16_t)(acc[v] + bc);
}

// ---- Kernel 4: scores[b,r,s] = (Q[r,:]·K[b,s,:]) / sqrt(E) ----
__global__ void __launch_bounds__(256)
scores_kernel(const bf16_t* __restrict__ Q, const bf16_t* __restrict__ Km,
              float* __restrict__ Sc_out) {
  int wave = (blockIdx.x * blockDim.x + threadIdx.x) >> 5;
  int nt = wave & 31, mt = (wave >> 5) & 7, b = wave >> 8;
  if (b >= Bc) return;
  v8f acc = {};
  const bf16_t* At = Q + (mt * 16) * Ec;
  const bf16_t* Bb = Km + ((size_t)b * Sc + nt * 16) * Ec;
#pragma unroll
  for (int kt = 0; kt < 4; ++kt) {
    v16bf a  = load_frag_a(At + kt * 32, Ec);
    v16bf bb = load_frag_b_rowK(Bb + kt * 32, Ec);
    acc = wmma_bf16(a, bb, acc);
  }
  const float scl = 0.08838834764831845f; // 1/sqrt(128)
  int lane = threadIdx.x & 31;
  int s = nt * 16 + (lane & 15);
  int rb = mt * 16 + ((lane & 16) ? 8 : 0);
  float* out = Sc_out + (((size_t)b * Rc + rb) * Sc) + s;
#pragma unroll
  for (int v = 0; v < 8; ++v) out[v * Sc] = acc[v] * scl;
}

// ---- Kernel 5: per-row exp(x - rowmax) -> bf16 (row scale cancels in ratio) ----
__global__ void __launch_bounds__(256)
softmax_e_kernel(const float* __restrict__ Sc_in, bf16_t* __restrict__ Eb) {
  int wave = (blockIdx.x * blockDim.x + threadIdx.x) >> 5;
  if (wave >= Bc * Rc) return;
  int lane = threadIdx.x & 31;
  const float* row = Sc_in + (size_t)wave * Sc;
  float vals[16];
  float mx = -3.0e38f;
#pragma unroll
  for (int i = 0; i < 16; ++i) { vals[i] = row[lane + i * 32]; mx = fmaxf(mx, vals[i]); }
#pragma unroll
  for (int off = 16; off >= 1; off >>= 1) mx = fmaxf(mx, __shfl_xor(mx, off, 32));
  bf16_t* er = Eb + (size_t)wave * Sc;
#pragma unroll
  for (int i = 0; i < 16; ++i) er[lane + i * 32] = (bf16_t)__expf(vals[i] - mx);
}

// ---- Kernel 6: enc = (E @ (x*mask)) / (E @ mask), bf16 out ----
__global__ void __launch_bounds__(256)
enc_kernel(const bf16_t* __restrict__ Em, const bf16_t* __restrict__ xmp,
           const bf16_t* __restrict__ mkp, bf16_t* __restrict__ enc) {
  int wave = (blockIdx.x * blockDim.x + threadIdx.x) >> 5;
  int nt = wave & 3, mt = (wave >> 2) & 7, b = wave >> 5;
  if (b >= Bc) return;
  v8f accN = {}, accD = {};
  const bf16_t* At = Em + ((size_t)b * Rc + mt * 16) * Sc;
  const bf16_t* Bx = xmp + (size_t)b * Sc * Dc;
  const bf16_t* Bm = mkp + (size_t)b * Sc * Dc;
#pragma unroll 4
  for (int kt = 0; kt < 16; ++kt) {
    v16bf a  = load_frag_a(At + kt * 32, Sc);
    v16bf bx = load_frag_b_pack(Bx + ((kt * 4 + nt) << 9));
    v16bf bm = load_frag_b_pack(Bm + ((kt * 4 + nt) << 9));
    accN = wmma_bf16(a, bx, accN);
    accD = wmma_bf16(a, bm, accD);
  }
  int lane = threadIdx.x & 31;
  int d = nt * 16 + (lane & 15);
  int rb = mt * 16 + ((lane & 16) ? 8 : 0);
#pragma unroll
  for (int v = 0; v < 8; ++v) {
    float den = accD[v];
    float val = (den > 1e-20f) ? (accN[v] / den) : 0.f;
    enc[((size_t)b * Rc + rb + v) * Dc + d] = (bf16_t)val;
  }
}

// ---- Kernel 7: gi = enc @ Wih^T (no bias), stored [t][b][3H] (bwd time-reversed) ----
__global__ void __launch_bounds__(256)
gi_kernel(const bf16_t* __restrict__ enc, const bf16_t* __restrict__ Wp,
          float* __restrict__ gi, int dir) {
  int wave = (blockIdx.x * blockDim.x + threadIdx.x) >> 5;
  int nt = wave % 48, mt = wave / 48;
  if (mt >= (Bc * Rc) / 16) return;
  v8f acc = {};
  const bf16_t* At = enc + (mt * 16) * Dc;
#pragma unroll
  for (int kt = 0; kt < 2; ++kt) {
    v16bf a = load_frag_a(At + kt * 32, Dc);
    v16bf b = load_frag_b_pack(Wp + ((kt * 48 + nt) << 9));
    acc = wmma_bf16(a, b, acc);
  }
  int lane = threadIdx.x & 31;
  int n = nt * 16 + (lane & 15);
  int m0 = mt * 16 + ((lane & 16) ? 8 : 0);
#pragma unroll
  for (int v = 0; v < 8; ++v) {
    int m = m0 + v;
    int b = m >> 7, r = m & 127;
    int t = dir ? (Rc - 1 - r) : r;
    gi[((size_t)t * Bc + b) * THc + n] = acc[v];
  }
}

// ---- Kernel 8: bidirectional GRU recurrence; 1 block/direction, 16 waves ----
__global__ void __launch_bounds__(512)
gru_kernel(const float* __restrict__ giAll,
           const bf16_t* __restrict__ WpF, const bf16_t* __restrict__ WpB,
           const float* __restrict__ bihF, const float* __restrict__ bhhF,
           const float* __restrict__ bihB, const float* __restrict__ bhhB,
           float* __restrict__ out) {
  int dir = blockIdx.x;
  const float* gi = giAll + (size_t)dir * Rc * Bc * THc;
  const bf16_t* Wp  = dir ? WpB  : WpF;
  const float*  bih = dir ? bihB : bihF;
  const float*  bhh = dir ? bhhB : bhhF;

  __shared__ __align__(16) bf16_t h_bf[Bc * Hc]; // 16 KB
  __shared__ float h_f[Bc * Hc];                 // 32 KB

  int tid = threadIdx.x;
  for (int i = tid; i < Bc * Hc; i += 512) { h_f[i] = 0.f; h_bf[i] = (bf16_t)0.f; }
  __syncthreads();

  int wave = tid >> 5;   // 0..15 == column group jg
  int lane = tid & 31;
  int jg = wave;
  int colL = jg * 16 + (lane & 15);

  float bih_r = bih[colL],       bhh_r = bhh[colL];
  float bih_z = bih[Hc + colL],  bhh_z = bhh[Hc + colL];
  float bih_n = bih[2*Hc + colL],bhh_n = bhh[2*Hc + colL];

  for (int t = 0; t < Rc; ++t) {
    v8f aR0 = {}, aR1 = {}, aZ0 = {}, aZ1 = {}, aN0 = {}, aN1 = {};
#pragma unroll 2
    for (int kt = 0; kt < 8; ++kt) {
      v16bf a0 = load_frag_a(h_bf + kt * 32, Hc);
      v16bf a1 = load_frag_a(h_bf + 16 * Hc + kt * 32, Hc);
      const bf16_t* bpR = Wp + ((kt * 48 + (jg))       << 9);
      const bf16_t* bpZ = Wp + ((kt * 48 + (16 + jg))  << 9);
      const bf16_t* bpN = Wp + ((kt * 48 + (32 + jg))  << 9);
      if (kt < 7) __builtin_prefetch(Wp + (((kt + 1) * 48 + jg) << 9), 0, 1);
      v16bf bR = load_frag_b_pack(bpR);
      v16bf bZ = load_frag_b_pack(bpZ);
      v16bf bN = load_frag_b_pack(bpN);
      aR0 = wmma_bf16(a0, bR, aR0);  aR1 = wmma_bf16(a1, bR, aR1);
      aZ0 = wmma_bf16(a0, bZ, aZ0);  aZ1 = wmma_bf16(a1, bZ, aZ1);
      aN0 = wmma_bf16(a0, bN, aN0);  aN1 = wmma_bf16(a1, bN, aN1);
    }
    __syncthreads(); // all reads of h done

    const float* git = gi + (size_t)t * Bc * THc;
    int rb = (lane & 16) ? 8 : 0;
    int tout = dir ? (Rc - 1 - t) : t;
#pragma unroll
    for (int v = 0; v < 8; ++v) {
#pragma unroll
      for (int half = 0; half < 2; ++half) {
        int brow = half * 16 + rb + v;
        float gR = (half ? aR1[v] : aR0[v]);
        float gZ = (half ? aZ1[v] : aZ0[v]);
        float gN = (half ? aN1[v] : aN0[v]);
        const float* gr = git + brow * THc;
        float r = 1.f / (1.f + __expf(-(gr[colL]        + bih_r + gR + bhh_r)));
        float z = 1.f / (1.f + __expf(-(gr[Hc + colL]   + bih_z + gZ + bhh_z)));
        float n = tanhf(gr[2*Hc + colL] + bih_n + r * (gN + bhh_n));
        float hp = h_f[brow * Hc + colL];
        float hn = (1.f - z) * n + z * hp;
        h_f[brow * Hc + colL] = hn;
        h_bf[brow * Hc + colL] = (bf16_t)hn;
        out[((size_t)brow * Rc + tout) * (2 * Hc) + dir * Hc + colL] = hn;
      }
    }
    __syncthreads(); // all writes of h done before next step's reads
  }
}

// ------------------------------ host launcher ------------------------------
extern "C" void kernel_launch(void* const* d_in, const int* in_sizes, int n_in,
                              void* d_out, int out_size, void* d_ws, size_t ws_size,
                              hipStream_t stream) {
  const float* x_vals     = (const float*)d_in[0];
  const int*   x_mask     = (const int*)  d_in[1];
  const float* time_steps = (const float*)d_in[2];
  const float* query_t    = (const float*)d_in[3];
  const float* Wq         = (const float*)d_in[4];
  const float* bq         = (const float*)d_in[5];
  const float* Wk         = (const float*)d_in[6];
  const float* bk         = (const float*)d_in[7];
  const float* w_per      = (const float*)d_in[8];
  const float* b_per      = (const float*)d_in[9];
  const float* w_lin      = (const float*)d_in[10];
  const float* b_lin      = (const float*)d_in[11];
  const float* Wih_f      = (const float*)d_in[12];
  const float* Whh_f      = (const float*)d_in[13];
  const float* bih_f      = (const float*)d_in[14];
  const float* bhh_f      = (const float*)d_in[15];
  const float* Wih_b      = (const float*)d_in[16];
  const float* Whh_b      = (const float*)d_in[17];
  const float* bih_b      = (const float*)d_in[18];
  const float* bhh_b      = (const float*)d_in[19];
  float* out = (float*)d_out;

  char* ws = (char*)d_ws;
  size_t off = 0;
  auto alloc = [&](size_t bytes) -> char* {
    char* p = ws + off;
    off = (off + bytes + 255) & ~(size_t)255;
    return p;
  };
  bf16_t* wq_pack   = (bf16_t*)alloc((size_t)Ec * Ec * 2);
  bf16_t* wk_pack   = (bf16_t*)alloc((size_t)Ec * Ec * 2);
  bf16_t* wihf_pack = (bf16_t*)alloc((size_t)THc * Dc * 2);
  bf16_t* wihb_pack = (bf16_t*)alloc((size_t)THc * Dc * 2);
  bf16_t* whhf_pack = (bf16_t*)alloc((size_t)THc * Hc * 2);
  bf16_t* whhb_pack = (bf16_t*)alloc((size_t)THc * Hc * 2);
  bf16_t* k_emb     = (bf16_t*)alloc((size_t)Bc * Sc * Ec * 2);
  bf16_t* q_emb     = (bf16_t*)alloc((size_t)Rc * Ec * 2);
  bf16_t* Kmat      = (bf16_t*)alloc((size_t)Bc * Sc * Ec * 2);
  bf16_t* Qmat      = (bf16_t*)alloc((size_t)Rc * Ec * 2);
  bf16_t* xm_pack   = (bf16_t*)alloc((size_t)Bc * Sc * Dc * 2);
  bf16_t* mk_pack   = (bf16_t*)alloc((size_t)Bc * Sc * Dc * 2);
  float*  scores    = (float*) alloc((size_t)Bc * Rc * Sc * 4);
  bf16_t* Emat      = (bf16_t*)alloc((size_t)Bc * Rc * Sc * 2);
  bf16_t* enc       = (bf16_t*)alloc((size_t)Bc * Rc * Dc * 2);
  float*  giAll     = (float*) alloc((size_t)2 * Rc * Bc * THc * 4);
  (void)ws_size; (void)n_in; (void)in_sizes; (void)out_size;

  auto cdiv = [](int a, int b) { return (a + b - 1) / b; };

  // 1) pack weights into B-fragment order
  pack_wT_kernel<<<cdiv(Ec * Ec, 256), 256, 0, stream>>>(Wq, wq_pack, Ec, Ec);
  pack_wT_kernel<<<cdiv(Ec * Ec, 256), 256, 0, stream>>>(Wk, wk_pack, Ec, Ec);
  pack_wT_kernel<<<cdiv(THc * Dc, 256), 256, 0, stream>>>(Wih_f, wihf_pack, THc, Dc);
  pack_wT_kernel<<<cdiv(THc * Dc, 256), 256, 0, stream>>>(Wih_b, wihb_pack, THc, Dc);
  pack_wT_kernel<<<cdiv(THc * Hc, 256), 256, 0, stream>>>(Whh_f, whhf_pack, THc, Hc);
  pack_wT_kernel<<<cdiv(THc * Hc, 256), 256, 0, stream>>>(Whh_b, whhb_pack, THc, Hc);

  // 2) time embeddings
  embed_kernel<<<cdiv((Bc * Sc + Rc) * Ec, 256), 256, 0, stream>>>(
      time_steps, query_t, w_per, b_per, w_lin, b_lin, k_emb, q_emb);

  // 3) pack x*mask / mask
  pack_x_kernel<<<cdiv(Bc * Sc * Dc, 256), 256, 0, stream>>>(x_vals, x_mask,
                                                             xm_pack, mk_pack);

  // 4) projections K = k_emb@Wk^T+bk, Q = q_emb@Wq^T+bq
  proj_kernel<<<(Bc * Sc) / 16, 256, 0, stream>>>(k_emb, wk_pack, bk, Kmat, Bc * Sc);
  proj_kernel<<<Rc / 16, 256, 0, stream>>>(q_emb, wq_pack, bq, Qmat, Rc);

  // 5) attention scores
  scores_kernel<<<(Bc * 8 * 32) / 8, 256, 0, stream>>>(Qmat, Kmat, scores);

  // 6) exp(x - rowmax)
  softmax_e_kernel<<<(Bc * Rc) / 8, 256, 0, stream>>>(scores, Emat);

  // 7) enc = (E @ xm) / (E @ mask)
  enc_kernel<<<(Bc * 8 * 4) / 8, 256, 0, stream>>>(Emat, xm_pack, mk_pack, enc);

  // 8) gi for both directions (backward stored time-reversed)
  {
    int waves = ((Bc * Rc) / 16) * 48;
    gi_kernel<<<waves / 8, 256, 0, stream>>>(enc, wihf_pack, giAll, 0);
    gi_kernel<<<waves / 8, 256, 0, stream>>>(enc, wihb_pack,
                                             giAll + (size_t)Rc * Bc * THc, 1);
  }

  // 9) GRU recurrence, one persistent workgroup per direction
  gru_kernel<<<2, 512, 0, stream>>>(giAll, whhf_pack, whhb_pack,
                                    bih_f, bhh_f, bih_b, bhh_b, out);
}
